// DRO_TOPK_78932908966340
// MI455X (gfx1250) — matrix operand order repaired and checked by reading, not compiled, not run on hardware
//
#include <hip/hip_runtime.h>
#include <hip/hip_bf16.h>

// Problem constants (from reference setup_inputs)
#define N_ROWS 4096
#define DIM    512
#define MARGIN 0.5f
#define BETA   0.0f
#define TOPK   20          // 4*K, K=5
#define TILE   128
#define NBX    (N_ROWS / TILE)   // 32
#define NBLK   (NBX * NBX)       // 1024
#define PART_STRIDE 25           // 20 top candidates + 5 partial sums per block
#define NTHREADS 256             // 8 wave32 waves per block

typedef _Float16 v16h __attribute__((ext_vector_type(16)));
typedef _Float16 v8h  __attribute__((ext_vector_type(8)));
typedef float    v8f  __attribute__((ext_vector_type(8)));

union V16U { v16h v; v8h h[2]; };

// ---------------------------------------------------------------------------
// Kernel 1: f32 -> f16 conversion (X is L2-resident; f16 halves WMMA operand
// traffic and enables v_wmma_f32_16x16x32_f16)
// ---------------------------------------------------------------------------
__global__ void __launch_bounds__(NTHREADS)
convert_f32_f16(const float* __restrict__ x, _Float16* __restrict__ xh, int n) {
    int i = blockIdx.x * blockDim.x + threadIdx.x;
    int stride = gridDim.x * blockDim.x;
    for (; i < n; i += stride) xh[i] = (_Float16)x[i];
}

// ---------------------------------------------------------------------------
// Device helper: thread-local sorted-descending top-20 insert (register file)
// ---------------------------------------------------------------------------
__device__ __forceinline__ void top_insert(float (&top)[TOPK], float v) {
    if (v > top[TOPK - 1]) {
#pragma unroll
        for (int j = 0; j < TOPK; ++j) {
            float t = top[j];
            bool gt = v > t;
            top[j] = gt ? v : t;
            v      = gt ? t : v;
        }
    }
}

// ---------------------------------------------------------------------------
// Kernel 2: fused 128x128 sim-tile GEMM (WMMA f16 -> f32) + loss epilogue.
// Each block emits 25 floats: top-20 losses of the tile + 5 partial sums
// (pos_sum, neg_sum, pos_cnt, neg_cnt, zero_cnt).
// ---------------------------------------------------------------------------
__global__ void __launch_bounds__(NTHREADS)
sim_tile_kernel(const _Float16* __restrict__ xh,
                const int*      __restrict__ targets,
                float*          __restrict__ partials)
{
    const int tid  = threadIdx.x;
    const int lane = tid & 31;
    const int wave = tid >> 5;           // 8 waves: each owns a 16x128 strip
    const int m    = lane & 15;
    const int hi   = lane >> 4;

    const int rowBlk  = blockIdx.y * TILE;
    const int colBlk  = blockIdx.x * TILE;
    const int rowBase = rowBlk + wave * 16;

    v8f acc[8] = {};                     // 8 col-tiles of 16x16 f32 accumulators

    // A-fragment base: lane holds row (rowBase+m), K runs at 8*hi and 8*hi+16
    const _Float16* aRow = xh + (size_t)(rowBase + m) * DIM + hi * 8;

    for (int kc = 0; kc < DIM; kc += 32) {
        V16U a;
        const v8h* pa = (const v8h*)(aRow + kc);
        a.h[0] = pa[0];                  // K = 8*hi .. +7
        a.h[1] = pa[2];                  // K = 8*hi+16 .. +23
#pragma unroll
        for (int t = 0; t < 8; ++t) {
            // B-fragment: lane holds column n=m of B == row (colBlk+16t+m) of X,
            // K = 8*hi..+7 and 8*hi+16..+23 (same order as A).
            V16U b;
            const v8h* pb = (const v8h*)(xh + (size_t)(colBlk + t * 16 + m) * DIM
                                            + (size_t)kc + hi * 8);
            b.h[0] = pb[0];
            b.h[1] = pb[2];
            acc[t] = __builtin_amdgcn_wmma_f32_16x16x32_f16(
                         false, a.v, false, b.v, (short)0, acc[t], false, false);
        }
    }

    // ---- epilogue: per-pair hinge loss, fused reductions ----
    int trow[8], tcol[8];
#pragma unroll
    for (int r = 0; r < 8; ++r) trow[r] = targets[rowBase + hi * 8 + r];
#pragma unroll
    for (int t = 0; t < 8; ++t) tcol[t] = targets[colBlk + t * 16 + m];

    float top[TOPK];
#pragma unroll
    for (int j = 0; j < TOPK; ++j) top[j] = -__builtin_inff();

    float pos_sum = 0.f, neg_sum = 0.f, pos_cnt = 0.f, neg_cnt = 0.f, zero_cnt = 0.f;

#pragma unroll
    for (int t = 0; t < 8; ++t) {
        const int col = colBlk + t * 16 + m;
#pragma unroll
        for (int r = 0; r < 8; ++r) {
            const int row = rowBase + hi * 8 + r;   // C layout: VGPR r, lane half
            if (row == col) continue;               // diagonal: invalid pair
            const float s = acc[t][r];
            float l;
            if (trow[r] == tcol[t]) {               // positive pair
                pos_sum += s; pos_cnt += 1.f;
                l = fmaxf(MARGIN + BETA - s, 0.f);
            } else {                                // negative pair
                neg_sum += s; neg_cnt += 1.f;
                l = fmaxf(MARGIN + s - BETA, 0.f);
            }
            if (l == 0.f) zero_cnt += 1.f;
            top_insert(top, l);
        }
    }

    // ---- deterministic block reductions in LDS ----
    __shared__ float red[NTHREADS];
    __shared__ float cand[NTHREADS * TOPK];
    __shared__ int   claim;

    float* blkOut = partials + (size_t)(blockIdx.y * NBX + blockIdx.x) * PART_STRIDE;

    float sums[5] = {pos_sum, neg_sum, pos_cnt, neg_cnt, zero_cnt};
    for (int i = 0; i < 5; ++i) {
        __syncthreads();
        red[tid] = sums[i];
        __syncthreads();
        for (int off = NTHREADS / 2; off > 0; off >>= 1) {
            if (tid < off) red[tid] += red[tid + off];
            __syncthreads();
        }
        if (tid == 0) blkOut[TOPK + i] = red[0];
    }

    // block top-20 merge: 20 rounds of max-extract over 256*20 candidates
    __syncthreads();
#pragma unroll
    for (int j = 0; j < TOPK; ++j) cand[tid * TOPK + j] = top[j];
    __syncthreads();

    for (int round = 0; round < TOPK; ++round) {
        float lm = -__builtin_inff();
#pragma unroll
        for (int j = 0; j < TOPK; ++j) lm = fmaxf(lm, cand[tid * TOPK + j]);
        red[tid] = lm;
        __syncthreads();
        for (int off = NTHREADS / 2; off > 0; off >>= 1) {
            if (tid < off) red[tid] = fmaxf(red[tid], red[tid + off]);
            __syncthreads();
        }
        const float gmax = red[0];
        if (tid == 0) { claim = 0; blkOut[round] = gmax; }
        __syncthreads();
        if (lm == gmax) {                    // remove exactly one instance
            if (atomicCAS(&claim, 0, 1) == 0) {
#pragma unroll
                for (int j = 0; j < TOPK; ++j) {
                    if (cand[tid * TOPK + j] == gmax) {
                        cand[tid * TOPK + j] = -__builtin_inff();
                        break;
                    }
                }
            }
        }
        __syncthreads();
    }
}

// ---------------------------------------------------------------------------
// Kernel 3: single-block finalize — merge 1024 block partials
// ---------------------------------------------------------------------------
__global__ void __launch_bounds__(NTHREADS)
finalize_kernel(const float* __restrict__ partials, float* __restrict__ out)
{
    const int tid = threadIdx.x;
    __shared__ float red[NTHREADS];
    __shared__ float cand[NTHREADS * TOPK];
    __shared__ int   claim;

    // --- sums (fixed-order, deterministic) ---
    float sums[5] = {0.f, 0.f, 0.f, 0.f, 0.f};
    for (int b = tid; b < NBLK; b += NTHREADS) {
        const float* p = partials + (size_t)b * PART_STRIDE + TOPK;
        sums[0] += p[0]; sums[1] += p[1]; sums[2] += p[2];
        sums[3] += p[3]; sums[4] += p[4];
    }
    float tot[5];
    for (int i = 0; i < 5; ++i) {
        __syncthreads();
        red[tid] = sums[i];
        __syncthreads();
        for (int off = NTHREADS / 2; off > 0; off >>= 1) {
            if (tid < off) red[tid] += red[tid + off];
            __syncthreads();
        }
        tot[i] = red[0];
    }

    // --- global top-20 over 1024*20 candidates ---
    float top[TOPK];
#pragma unroll
    for (int j = 0; j < TOPK; ++j) top[j] = -__builtin_inff();
    for (int idx = tid; idx < NBLK * TOPK; idx += NTHREADS) {
        const int b = idx / TOPK, j = idx % TOPK;
        top_insert(top, partials[(size_t)b * PART_STRIDE + j]);
    }

    __syncthreads();
#pragma unroll
    for (int j = 0; j < TOPK; ++j) cand[tid * TOPK + j] = top[j];
    __syncthreads();

    float loss_sum = 0.f;
    for (int round = 0; round < TOPK; ++round) {
        float lm = -__builtin_inff();
#pragma unroll
        for (int j = 0; j < TOPK; ++j) lm = fmaxf(lm, cand[tid * TOPK + j]);
        red[tid] = lm;
        __syncthreads();
        for (int off = NTHREADS / 2; off > 0; off >>= 1) {
            if (tid < off) red[tid] = fmaxf(red[tid], red[tid + off]);
            __syncthreads();
        }
        const float gmax = red[0];
        loss_sum += gmax;
        if (tid == 0) claim = 0;
        __syncthreads();
        if (lm == gmax) {
            if (atomicCAS(&claim, 0, 1) == 0) {
#pragma unroll
                for (int j = 0; j < TOPK; ++j) {
                    if (cand[tid * TOPK + j] == gmax) {
                        cand[tid * TOPK + j] = -__builtin_inff();
                        break;
                    }
                }
            }
        }
        __syncthreads();
    }

    if (tid == 0) {
        out[0] = loss_sum / (float)TOPK;   // mean of top-(4K) losses
        out[1] = tot[4];                   // num_of_zeros
        out[2] = tot[0] / tot[2];          // sum(pos sim)/count(pos)
        out[3] = tot[1] / tot[3];          // sum(neg sim)/count(neg)
    }
}

// ---------------------------------------------------------------------------
// Workspace layout: [0, 4MB) f16 copy of X; [4MB, ...) 1024*25 partial floats
// ---------------------------------------------------------------------------
extern "C" void kernel_launch(void* const* d_in, const int* in_sizes, int n_in,
                              void* d_out, int out_size, void* d_ws, size_t ws_size,
                              hipStream_t stream) {
    const float* x       = (const float*)d_in[0];
    const int*   targets = (const int*)d_in[1];
    float*       out     = (float*)d_out;

    _Float16* xh = (_Float16*)d_ws;
    float* partials = (float*)((char*)d_ws + (size_t)N_ROWS * DIM * sizeof(_Float16));

    convert_f32_f16<<<512, NTHREADS, 0, stream>>>(x, xh, N_ROWS * DIM);

    dim3 grid(NBX, NBX);
    sim_tile_kernel<<<grid, NTHREADS, 0, stream>>>(xh, targets, partials);

    finalize_kernel<<<1, NTHREADS, 0, stream>>>(partials, out);
}
